// RetinaNet_21174188769317
// MI455X (gfx1250) — matrix operand order repaired and checked by reading, not compiled, not run on hardware
//
#include <hip/hip_runtime.h>
#include <hip/hip_bf16.h>

typedef __attribute__((ext_vector_type(16))) __bf16 v16bf;
typedef __attribute__((ext_vector_type(8)))  __bf16 v8bf;
typedef __attribute__((ext_vector_type(8)))  float  v8f;

#define CIN 256

// ---------------------------------------------------------------------------
// Weight repack: fp32 W[O][I][3][3] -> bf16 wpk[tap][kchunk32][c][k]
//   tap = kh*3+kw, kchunk32 = i/32, k = i%32, c = o (zero-padded to CoutPad)
// B-fragment for lane (n=lane%16, half=lane/16) is 16 contiguous bf16 at
// [... c*32 + 16*half] -- contiguous k order, same as the A fragment (lanes0-15 hold K=0-15).
// ---------------------------------------------------------------------------
__global__ __launch_bounds__(256) void pack_w_k(const float* __restrict__ w,
                                                __bf16* __restrict__ out,
                                                int Cout, int CoutPad) {
  long idx = (long)blockIdx.x * 256 + threadIdx.x;
  long total = (long)9 * 8 * CoutPad * 32;
  if (idx >= total) return;
  int k = (int)(idx & 31);
  long t = idx >> 5;
  int c  = (int)(t % CoutPad); t /= CoutPad;
  int ch = (int)(t & 7);
  int tap = (int)(t >> 3);
  float v = 0.f;
  if (c < Cout) v = w[((long)c * CIN + ch * 32 + k) * 9 + tap];
  out[idx] = (__bf16)v;
}

// fp32 NCHW [4,256,H,W] -> bf16 NHWC rows [M=4*H*W][256]
__global__ __launch_bounds__(256) void cvt_k(const float* __restrict__ in,
                                             __bf16* __restrict__ out, int HW) {
  long idx = (long)blockIdx.x * 256 + threadIdx.x;
  long total = 4L * HW * CIN;
  if (idx >= total) return;
  int c = (int)(idx & 255);
  long m = idx >> 8;
  int n  = (int)(m / HW);
  int hw = (int)(m % HW);
  out[idx] = (__bf16)in[((long)n * CIN + c) * HW + hw];
}

// ---------------------------------------------------------------------------
// Implicit-GEMM conv3x3 via V_WMMA_F32_16X16X32_BF16.
// Block: 256 threads = 8 waves; tile 128 M (spatial) x 64 N (out-chan).
// Stages: 9 taps x 4 chunks of 64 input channels = 36; double-buffered LDS,
// one barrier per stage, 8 WMMAs per wave per stage.
// Weights staged with GLOBAL_LOAD_ASYNC_TO_LDS_B128 (ASYNCcnt); activations
// prefetched to SSA uint4 regs one stage ahead (zero-filled SAME padding).
// ---------------------------------------------------------------------------
template <bool FINAL>
__global__ __launch_bounds__(256)
void conv3x3_wmma(const __bf16* __restrict__ in,   // [M][256] bf16 NHWC
                  const __bf16* __restrict__ wpk,  // [9][8][CoutPad][32] bf16
                  const float* __restrict__ bias,  // [Cout]
                  void* __restrict__ outp,
                  int H, int W, int M, int CoutPad, int Cout,
                  long out_base, long stride_n) {
  // sA[buf][ksub][row 0..127][k 0..31], sB[buf][ksub][c 0..63][k 0..31]
  __shared__ __align__(64) __bf16 sA[2 * 2 * 128 * 32];  // 32 KB
  __shared__ __align__(64) __bf16 sB[2 * 2 * 64 * 32];   // 16 KB

  const int tid  = threadIdx.x;
  const int m0   = blockIdx.x * 128;
  const int n0   = blockIdx.y * 64;
  const int wv   = tid >> 5;
  const int lane = tid & 31;
  const int half = lane >> 4;
  const int lm   = lane & 15;
  const int HW   = H * W;

  // A staging: 2 threads per spatial row; each covers one 32-ch sub-chunk.
  const int arow  = tid >> 1;
  const int ahalf = tid & 1;
  const int am    = m0 + arow;
  const bool avalid = am < M;
  int an = 0, ah = 0, aw = 0;
  if (avalid) { an = am / HW; int hw = am % HW; ah = hw / W; aw = hw % W; }

  // B staging (async): 128 threads per 4 KB sub-chunk, 32 B each.
  const int bsub   = tid >> 7;
  const int binner = tid & 127;
  const unsigned sB0 = (unsigned)(unsigned long long)(void*)&sB[0];
  const unsigned ldsB_lane = sB0 + (unsigned)((bsub * 2048 + binner * 16) * 2);

  v8f acc[4];
  for (int nt = 0; nt < 4; ++nt)
    for (int e = 0; e < 8; ++e) acc[nt][e] = 0.f;

  // A prefetch registers: scalars so they stay SSA (never hit scratch).
  uint4 a0, a1, a2, a3;

#define LOAD_A(S)                                                              \
  {                                                                            \
    const int tap_ = (S) >> 2, kc_ = (S) & 3;                                  \
    const int hs_ = ah + tap_ / 3 - 1;                                         \
    const int ws_ = aw + tap_ % 3 - 1;                                         \
    const bool inb_ =                                                          \
        avalid && hs_ >= 0 && hs_ < H && ws_ >= 0 && ws_ < W;                  \
    uint4 z_ = {0, 0, 0, 0};                                                   \
    a0 = z_; a1 = z_; a2 = z_; a3 = z_;                                        \
    if (inb_) {                                                                \
      const uint4* gp_ = (const uint4*)(in +                                   \
          ((long)(an * H + hs_) * W + ws_) * CIN + kc_ * 64 + ahalf * 32);     \
      a0 = gp_[0]; a1 = gp_[1]; a2 = gp_[2]; a3 = gp_[3];                      \
    }                                                                          \
  }

#define LOAD_B(S, BUF)                                                         \
  {                                                                            \
    const int tap_ = (S) >> 2, kc_ = (S) & 3;                                  \
    const __bf16* gp_ = wpk +                                                  \
        (((long)tap_ * 8 + kc_ * 2 + bsub) * CoutPad + n0) * 32 + binner * 16; \
    const unsigned lds_ = ldsB_lane + (unsigned)((BUF) * 8192);                \
    asm volatile("global_load_async_to_lds_b128 %0, %1, off" ::"v"(lds_),      \
                 "v"((unsigned long long)gp_));                                \
    asm volatile("global_load_async_to_lds_b128 %0, %1, off offset:16" ::"v"(  \
                     lds_),                                                    \
                 "v"((unsigned long long)gp_));                                \
  }

  // Pipeline: stage s+1 issued after barrier(s); buffer (s+1)&1 was last read
  // at compute(s-1), which all waves completed before barrier(s).
  LOAD_A(0);
  LOAD_B(0, 0);
  for (int s = 0; s < 36; ++s) {
    const int buf = s & 1;
    {
      uint4* p = (uint4*)&sA[(((buf * 2 + ahalf) * 128) + arow) * 32];
      p[0] = a0; p[1] = a1; p[2] = a2; p[3] = a3;
    }
    asm volatile("s_wait_asynccnt 0x0" ::: "memory");  // B(buf) resident
    __syncthreads();
    if (s + 1 < 36) {
      LOAD_A(s + 1);
      LOAD_B(s + 1, buf ^ 1);
    }
    // compute stage s: 2 k-sub-chunks x 4 n-tiles = 8 WMMAs
    for (int ks = 0; ks < 2; ++ks) {
      const int abase = ((buf * 2 + ks) * 128 + wv * 16 + lm) * 32;
      // A fragment: element e -> k = 16*half + e (contiguous, matches B)
      v8bf alo = *(const v8bf*)&sA[abase + half * 16];
      v8bf ahi = *(const v8bf*)&sA[abase + half * 16 + 8];
      v16bf afrag;
      for (int e = 0; e < 8; ++e) { afrag[e] = alo[e]; afrag[e + 8] = ahi[e]; }
      for (int nt = 0; nt < 4; ++nt) {
        // B fragment: lane n = lm, k = 16*half + e -> contiguous in sB
        v16bf bfrag = *(const v16bf*)
            &sB[((buf * 2 + ks) * 64 + nt * 16 + lm) * 32 + half * 16];
        acc[nt] = __builtin_amdgcn_wmma_f32_16x16x32_bf16(
            false, afrag, false, bfrag, (short)0, acc[nt], false, false);
      }
    }
  }
#undef LOAD_A
#undef LOAD_B

  // Epilogue. D layout: lane col = lm, VGPR r -> local row = r + 8*half.
  for (int nt = 0; nt < 4; ++nt) {
    const int col = n0 + nt * 16 + lm;
    if (col >= Cout) continue;
    const float bv = bias[col];
    for (int r = 0; r < 8; ++r) {
      const int mg = m0 + wv * 16 + half * 8 + r;
      if (mg >= M) continue;
      float v = acc[nt][r] + bv;
      if (!FINAL) {
        v = v > 0.f ? v : 0.f;
        ((__bf16*)outp)[(long)mg * CIN + col] = (__bf16)v;
      } else {
        const int n  = mg / HW;
        const int hw = mg % HW;
        ((float*)outp)[out_base + (long)n * stride_n + (long)hw * Cout + col] = v;
      }
    }
  }
}

// ---------------------------------------------------------------------------
extern "C" void kernel_launch(void* const* d_in, const int* in_sizes, int n_in,
                              void* d_out, int out_size, void* d_ws,
                              size_t ws_size, hipStream_t stream) {
  (void)in_sizes; (void)n_in; (void)out_size; (void)ws_size;

  static const int Hs[5] = {100, 50, 25, 13, 7};
  const float* feats[5];
  for (int i = 0; i < 5; ++i) feats[i] = (const float*)d_in[i];
  const float* cls_w     = (const float*)d_in[5];
  const float* cls_b     = (const float*)d_in[6];
  const float* cls_out_w = (const float*)d_in[7];
  const float* cls_out_b = (const float*)d_in[8];
  const float* reg_w     = (const float*)d_in[9];
  const float* reg_b     = (const float*)d_in[10];
  const float* reg_out_w = (const float*)d_in[11];
  const float* reg_out_b = (const float*)d_in[12];
  float* out = (float*)d_out;

  // Workspace carve-out
  char* ws = (char*)d_ws;
  size_t off = 0;
  auto carve = [&](size_t bytes) -> void* {
    void* p = ws + off;
    off += (bytes + 255) & ~(size_t)255;
    return p;
  };
  const size_t WL = (size_t)9 * 8 * 256 * 32;  // elems per inner-layer pack
  __bf16* clsw_pk[4];
  __bf16* regw_pk[4];
  for (int i = 0; i < 4; ++i) clsw_pk[i] = (__bf16*)carve(WL * 2);
  for (int i = 0; i < 4; ++i) regw_pk[i] = (__bf16*)carve(WL * 2);
  const size_t WCO = (size_t)9 * 8 * 832 * 32;  // cls final (819 -> pad 832)
  const size_t WRO = (size_t)9 * 8 * 64 * 32;   // reg final (36 -> pad 64)
  __bf16* cls_out_pk = (__bf16*)carve(WCO * 2);
  __bf16* reg_out_pk = (__bf16*)carve(WRO * 2);
  const size_t ACT = (size_t)4 * 100 * 100 * 256;  // largest level
  __bf16* bufA = (__bf16*)carve(ACT * 2);
  __bf16* bufB = (__bf16*)carve(ACT * 2);

  // ---- pack weights ----
  {
    int nb = (int)((WL + 255) / 256);
    for (int i = 0; i < 4; ++i) {
      pack_w_k<<<nb, 256, 0, stream>>>(cls_w + (size_t)i * 256 * 256 * 9,
                                       clsw_pk[i], 256, 256);
      pack_w_k<<<nb, 256, 0, stream>>>(reg_w + (size_t)i * 256 * 256 * 9,
                                       regw_pk[i], 256, 256);
    }
    pack_w_k<<<(int)((WCO + 255) / 256), 256, 0, stream>>>(cls_out_w,
                                                           cls_out_pk, 819, 832);
    pack_w_k<<<(int)((WRO + 255) / 256), 256, 0, stream>>>(reg_out_w,
                                                           reg_out_pk, 36, 64);
  }

  const long TOTPOS    = 13343;                  // sum H*W over levels
  const long CLS_TOTAL = 4L * TOTPOS * 9 * 91;   // cls elems before reg block
  long cum = 0;  // cumulative H*W of previous levels

  for (int lv = 0; lv < 5; ++lv) {
    const int H = Hs[lv], W = Hs[lv];
    const int HW = H * W;
    const int M = 4 * HW;
    const long cvt_total = (long)M * 256;
    const int cvt_nb = (int)((cvt_total + 255) / 256);
    dim3 g((M + 127) / 128, 4);       // inner convs: 256 cols
    dim3 gc((M + 127) / 128, 13);     // cls final: 832 cols
    dim3 gr((M + 127) / 128, 1);      // reg final: 64 cols

    // ---- cls branch ----
    cvt_k<<<cvt_nb, 256, 0, stream>>>(feats[lv], bufA, HW);
    conv3x3_wmma<false><<<g, 256, 0, stream>>>(bufA, clsw_pk[0], cls_b + 0,
                                               bufB, H, W, M, 256, 256, 0, 0);
    conv3x3_wmma<false><<<g, 256, 0, stream>>>(bufB, clsw_pk[1], cls_b + 256,
                                               bufA, H, W, M, 256, 256, 0, 0);
    conv3x3_wmma<false><<<g, 256, 0, stream>>>(bufA, clsw_pk[2], cls_b + 512,
                                               bufB, H, W, M, 256, 256, 0, 0);
    conv3x3_wmma<false><<<g, 256, 0, stream>>>(bufB, clsw_pk[3], cls_b + 768,
                                               bufA, H, W, M, 256, 256, 0, 0);
    conv3x3_wmma<true><<<gc, 256, 0, stream>>>(
        bufA, cls_out_pk, cls_out_b, out, H, W, M, 832, 819,
        /*out_base=*/819L * cum, /*stride_n=*/TOTPOS * 9L * 91);

    // ---- reg branch ----
    cvt_k<<<cvt_nb, 256, 0, stream>>>(feats[lv], bufB, HW);
    conv3x3_wmma<false><<<g, 256, 0, stream>>>(bufB, regw_pk[0], reg_b + 0,
                                               bufA, H, W, M, 256, 256, 0, 0);
    conv3x3_wmma<false><<<g, 256, 0, stream>>>(bufA, regw_pk[1], reg_b + 256,
                                               bufB, H, W, M, 256, 256, 0, 0);
    conv3x3_wmma<false><<<g, 256, 0, stream>>>(bufB, regw_pk[2], reg_b + 512,
                                               bufA, H, W, M, 256, 256, 0, 0);
    conv3x3_wmma<false><<<g, 256, 0, stream>>>(bufA, regw_pk[3], reg_b + 768,
                                               bufB, H, W, M, 256, 256, 0, 0);
    conv3x3_wmma<true><<<gr, 256, 0, stream>>>(
        bufB, reg_out_pk, reg_out_b, out, H, W, M, 64, 36,
        /*out_base=*/CLS_TOTAL + 36L * cum, /*stride_n=*/TOTPOS * 9L * 4);

    cum += HW;
  }
}
